// NEGA_79998060855418
// MI455X (gfx1250) — matrix operand order repaired and checked by reading, not compiled, not run on hardware
//
#include <hip/hip_runtime.h>
#include <hip/hip_fp16.h>

// ---------------------------------------------------------------------------
// GraphRec-style forward, CDNA5 (gfx1250, wave32).  All dense GEMMs run on
// v_wmma_f32_16x16x32_f16.  Fragments are built directly in registers:
//  - weights pre-converted once to transposed f16 (Wt[N][K]) -> 2x b128/lane
//  - A operand composed on the fly from fp32 sources -> 4x b128/lane + cvt
// No LDS tiles, no barriers in the K-loop.
// ---------------------------------------------------------------------------

typedef _Float16 v16h __attribute__((ext_vector_type(16)));
typedef _Float16 v8h  __attribute__((ext_vector_type(8)));
typedef float    v8f  __attribute__((ext_vector_type(8)));

#define DEVI __device__ __forceinline__

constexpr int BATCH = 2048;
constexpr int KNB   = 64;          // fixed neighbor degree
constexpr int MROWS = BATCH * KNB; // 131072
constexpr float EPS_BN = 1e-5f;

DEVI float sigmoidf_(float x) { return 1.f / (1.f + expf(-x)); }
DEVI float seluf_(float x) {
  const float L = 1.0507009873554805f, A = 1.6732632423543772f;
  return x > 0.f ? L * x : L * A * (expf(x) - 1.f);
}
DEVI float safe_powf_(float z, float p) {
  return z > 0.f ? exp2f(p * log2f(z)) : 0.f;
}
DEVI float wave_sum(float v) {
#pragma unroll
  for (int o = 16; o > 0; o >>= 1) v += __shfl_xor(v, o, 32);
  return v;
}
DEVI float wave_max(float v) {
#pragma unroll
  for (int o = 16; o > 0; o >>= 1) v = fmaxf(v, __shfl_xor(v, o, 32));
  return v;
}

DEVI v8h load8f16(const float* p) {       // 8 contiguous f32 -> v8h
  float4 a = ((const float4*)p)[0];
  float4 b = ((const float4*)p)[1];
  v8h r;
  r[0] = (_Float16)a.x; r[1] = (_Float16)a.y;
  r[2] = (_Float16)a.z; r[3] = (_Float16)a.w;
  r[4] = (_Float16)b.x; r[5] = (_Float16)b.y;
  r[6] = (_Float16)b.z; r[7] = (_Float16)b.w;
  return r;
}
DEVI v8h load8prod(const float* p, const float* q) {  // (p*q) -> v8h
  float4 a = ((const float4*)p)[0], b = ((const float4*)p)[1];
  float4 c = ((const float4*)q)[0], d = ((const float4*)q)[1];
  v8h r;
  r[0] = (_Float16)(a.x * c.x); r[1] = (_Float16)(a.y * c.y);
  r[2] = (_Float16)(a.z * c.z); r[3] = (_Float16)(a.w * c.w);
  r[4] = (_Float16)(b.x * d.x); r[5] = (_Float16)(b.y * d.y);
  r[6] = (_Float16)(b.z * d.z); r[7] = (_Float16)(b.w * d.w);
  return r;
}

// ------------------------- WMMA GEMM ---------------------------------------
enum { A_PLAIN = 0, A_CONCAT_BCAST = 1, A_CONCAT3 = 2, A_GATE_GATHER = 3 };
enum { E_NONE = 0, E_SELU = 1, E_SIGMOID = 2, E_GATE_O = 3, E_COMB = 4 };

// 8 contiguous A elements starting at kk (8-aligned, within one 128-segment).
template <int MODE, int KTOT>
DEVI v8h fetchA8(const float* A0, const float* A1, const float* embT,
                 const float* embR, int m, int kk, int itA, int rrA) {
  if (MODE == A_PLAIN) {
    return load8f16(A0 + (long)m * KTOT + kk);
  } else if (MODE == A_CONCAT_BCAST) {   // [o_n(m) | rep(m>>6)]
    return (kk < 128) ? load8f16(A0 + (long)m * 128 + kk)
                      : load8f16(A1 + (long)(m >> 6) * 128 + (kk - 128));
  } else if (MODE == A_CONCAT3) {        // [S | T | S*T]
    if (kk < 128)      return load8f16(A0 + (long)m * 128 + kk);
    else if (kk < 256) return load8f16(A1 + (long)m * 128 + (kk - 128));
    int d = kk - 256;
    return load8prod(A0 + (long)m * 128 + d, A1 + (long)m * 128 + d);
  } else {                                // A_GATE_GATHER: [e_ui|e_r|e_ui*e_r]
    if (kk < 128)      return load8f16(embT + (long)itA * 128 + kk);
    else if (kk < 256) return load8f16(embR + (long)rrA * 128 + (kk - 128));
    int d = kk - 256;
    return load8prod(embT + (long)itA * 128 + d, embR + (long)rrA * 128 + d);
  }
}

// C[M,N] = epi( A_virtual[M,KTOT] @ W + bias ), Wt is f16 transposed [N][KTOT].
// 128 threads = 4 waves; each wave computes one 16x16 tile.
template <int MODE, int EPI, int KTOT>
__global__ __launch_bounds__(128) void gemm_wmma_kernel(
    const float* __restrict__ A0, const float* __restrict__ A1,
    const _Float16* __restrict__ Wt, const float* __restrict__ bias,
    float* __restrict__ Cout,
    const int* __restrict__ nodes, const int* __restrict__ idxTab,
    const int* __restrict__ ratTab, const float* __restrict__ embT,
    const float* __restrict__ embR, const float* __restrict__ Sg,
    const float* __restrict__ Tg, int M, int N) {
  __shared__ int ldsIt[64];
  __shared__ int ldsR[64];

  const int tid   = threadIdx.x;
  const int lane  = tid & 31;
  const int wv    = tid >> 5;
  const int mBase = blockIdx.x * 64;
  const int m0    = mBase + wv * 16;
  const int n0    = blockIdx.y * 16;
  const int col   = lane & 15;   // output column within tile, == A row index
  const int kh    = lane >> 4;

  if (MODE == A_GATE_GATHER) {
    if (tid < 64) {
      int m = mBase + tid;
      int u = nodes[m >> 6];
      ldsIt[tid] = idxTab[u * KNB + (m & 63)];
      ldsR[tid]  = ratTab[u * KNB + (m & 63)];
    }
    __syncthreads();
  }

  const int mA = m0 + col;       // this lane's A-matrix row
  int itA = 0, rrA = 0;
  if (MODE == A_GATE_GATHER) {   // hoist gather indices out of the K-loop
    itA = ldsIt[mA - mBase];
    rrA = ldsR[mA - mBase];
  }
  const _Float16* wrow = Wt + (long)(n0 + col) * KTOT;

  v8f acc = {};
#pragma unroll
  for (int k0 = 0; k0 < KTOT; k0 += 32) {
    // A fragment: two contiguous 8-element K-groups of row mA
    v8h alo = fetchA8<MODE, KTOT>(A0, A1, embT, embR, mA, k0 + kh * 8, itA, rrA);
    v8h ahi = fetchA8<MODE, KTOT>(A0, A1, embT, embR, mA, k0 + 16 + kh * 8, itA, rrA);
    // B fragment: two 8-element K-groups of Wt row (n0+col), matching A
    const v8h* pw = (const v8h*)(wrow + k0);
    v8h blo = pw[kh];
    v8h bhi = pw[2 + kh];
    if (k0 + 32 < KTOT) __builtin_prefetch(wrow + k0 + 32, 0, 3);
    v16h av, bv;
#pragma unroll
    for (int i = 0; i < 8; ++i) {
      av[i] = alo[i]; av[i + 8] = ahi[i];
      bv[i] = blo[i]; bv[i + 8] = bhi[i];
    }
    acc = __builtin_amdgcn_wmma_f32_16x16x32_f16(false, av, false, bv,
                                                 (short)0, acc, false, false);
  }

  // ---- epilogue: lane -> (col, rows m0 + v + 8*kh)
  const int n = n0 + col;
  const float bb = bias ? bias[n] : 0.f;
#pragma unroll
  for (int v = 0; v < 8; ++v) {
    int m = m0 + v + kh * 8;
    float x = acc[v] + bb;
    float outv;
    if (EPI == E_NONE)         outv = x;
    else if (EPI == E_SELU)    outv = seluf_(x);
    else if (EPI == E_SIGMOID) outv = sigmoidf_(x);
    else if (EPI == E_GATE_O) {              // o = a*e_ui + (1-a)*e_r
      int lr = m - mBase;
      int it = ldsIt[lr], rr = ldsR[lr];
      float e1 = embT[(long)it * 128 + n];
      float e2 = embR[(long)rr * 128 + n];
      float g = sigmoidf_(x);
      outv = g * e1 + (1.f - g) * e2;
    } else {                                  // E_COMB: g*S + (1-g)*T
      float g = sigmoidf_(x);
      outv = g * Sg[(long)m * 128 + n] + (1.f - g) * Tg[(long)m * 128 + n];
    }
    Cout[(long)m * N + n] = outv;
  }
}

// ---- weight pre-pass: W[K][N] f32 -> Wt[N][K] f16 (runs once per launch)
__global__ __launch_bounds__(256) void wtrans_kernel(
    const float* __restrict__ src, _Float16* __restrict__ dst, int K, int N) {
  int i = blockIdx.x * 256 + threadIdx.x;
  if (i >= K * N) return;
  int n = i / K, k = i - n * K;
  dst[(long)n * K + k] = (_Float16)src[(long)k * N + n];
}

// ------------------------- helper kernels ----------------------------------
template <int CD, int EPID>   // EPID: 0 none, 1 sigmoid+1
__global__ __launch_bounds__(128) void rowdot_kernel(
    const float* __restrict__ X, const float* __restrict__ w,
    const float* __restrict__ b, float* __restrict__ y, int R) {
  int row  = blockIdx.x * 4 + (threadIdx.x >> 5);
  int lane = threadIdx.x & 31;
  if (row >= R) return;
  float s = 0.f;
#pragma unroll
  for (int d = lane; d < CD; d += 32) s += X[(long)row * CD + d] * w[d];
  s = wave_sum(s);
  if (lane == 0) {
    float v = s + b[0];
    if (EPID == 1) v = sigmoidf_(v) + 1.f;
    y[row] = v;
  }
}

__global__ __launch_bounds__(128) void l2norm_kernel(
    const float* __restrict__ X, float* __restrict__ Y) {
  long row = blockIdx.x;
  int t = threadIdx.x;
  float v = X[row * 128 + t];
  __shared__ float red[4];
  float s = wave_sum(v * v);
  if ((t & 31) == 0) red[t >> 5] = s;
  __syncthreads();
  float tot = red[0] + red[1] + red[2] + red[3];
  float scale = 1.f / fmaxf(sqrtf(tot), 1e-12f);
  Y[row * 128 + t] = v * scale;
}

__global__ __launch_bounds__(128) void gather_rows_kernel(
    const int* __restrict__ nodes, const int* __restrict__ nbr,
    const float* __restrict__ emb, float* __restrict__ O) {
  long m = blockIdx.x;
  int t = threadIdx.x;
  int b = (int)(m >> 6), k = (int)(m & 63);
  int src = nbr[nodes[b] * KNB + k];
  O[m * 128 + t] = emb[(long)src * 128 + t];
}

__global__ __launch_bounds__(128) void gather_self_kernel(
    const int* __restrict__ nodes, const float* __restrict__ emb,
    float* __restrict__ Y) {
  long b = blockIdx.x;
  int t = threadIdx.x;
  Y[b * 128 + t] = emb[(long)nodes[b] * 128 + t];
}

__global__ __launch_bounds__(32) void entmax_kernel(
    const float* __restrict__ S, const float* __restrict__ Al,
    float* __restrict__ Wout) {
  long b = blockIdx.x;
  int lane = threadIdx.x;
  long m0 = b * 64;
  float x0 = S[m0 + lane],  x1 = S[m0 + lane + 32];
  float a0 = Al[m0 + lane], a1 = Al[m0 + lane + 32];
  float am0 = a0 - 1.f, am1 = a1 - 1.f;
  float xs0 = x0 * am0, xs1 = x1 * am1;
  float iv0 = 1.f / am0, iv1 = 1.f / am1;
  float tau = wave_max(fmaxf(xs0, xs1)) - 1.f;
  float f_lo =
      wave_sum(safe_powf_(xs0 - tau, iv0) + safe_powf_(xs1 - tau, iv1)) - 1.f;
  float dm = 1.f;
#pragma unroll 1
  for (int it = 0; it < 30; ++it) {
    dm *= 0.5f;
    float tm = tau + dm;
    float fm =
        wave_sum(safe_powf_(xs0 - tm, iv0) + safe_powf_(xs1 - tm, iv1)) - 1.f;
    if (fm * f_lo >= 0.f) tau = tm;   // uniform across wave (post-reduction)
  }
  float p0 = safe_powf_(xs0 - tau, iv0), p1 = safe_powf_(xs1 - tau, iv1);
  float invt = 1.f / wave_sum(p0 + p1);
  Wout[b * 64 + lane]      = p0 * invt;
  Wout[b * 64 + lane + 32] = p1 * invt;
}

__global__ __launch_bounds__(128) void wsum_kernel(
    const float* __restrict__ Wt, const float* __restrict__ On,
    float* __restrict__ repRaw, float* __restrict__ feats, int init) {
  long b = blockIdx.x;
  int t = threadIdx.x;
  __shared__ float wsh[64];
  if (t < 64) wsh[t] = Wt[b * 64 + t];
  __syncthreads();
  float acc = 0.f;
#pragma unroll 4
  for (int k = 0; k < 64; ++k)
    acc += wsh[k] * On[(b * 64 + k) * 128 + t];
  repRaw[b * 128 + t] = acc;
  if (init) feats[b * 128 + t] = acc;
  else      feats[b * 128 + t] += acc;
}

__global__ __launch_bounds__(256) void bn_stats_kernel(
    const float* __restrict__ X, float* __restrict__ mu,
    float* __restrict__ var, int R, int C, float prescale) {
  int c = blockIdx.x;
  int t = threadIdx.x;
  float s = 0.f, s2 = 0.f;
  for (int r = t; r < R; r += 256) {
    float v = X[(long)r * C + c] * prescale;
    s += v; s2 += v * v;
  }
  __shared__ float sh[256], sh2[256];
  sh[t] = s; sh2[t] = s2;
  __syncthreads();
  for (int o = 128; o > 0; o >>= 1) {
    if (t < o) { sh[t] += sh[t + o]; sh2[t] += sh2[t + o]; }
    __syncthreads();
  }
  if (t == 0) {
    float m = sh[0] / R;
    mu[c] = m;
    var[c] = sh2[0] / R - m * m;
  }
}

__global__ __launch_bounds__(256) void bn_apply_kernel(
    const float* __restrict__ X, float* __restrict__ Y,
    const float* __restrict__ mu, const float* __restrict__ var,
    const float* __restrict__ g, const float* __restrict__ bb, long total,
    int C, float prescale) {
  long i = (long)blockIdx.x * 256 + threadIdx.x;
  if (i >= total) return;
  int c = (int)(i % C);
  float v = X[i] * prescale;
  Y[i] = (v - mu[c]) * rsqrtf(var[c] + EPS_BN) * g[c] + bb[c];
}

__global__ __launch_bounds__(256) void concat2_kernel(
    const float* __restrict__ A, const float* __restrict__ B,
    float* __restrict__ Y) {
  long b = blockIdx.x;
  int t = threadIdx.x;
  Y[b * 256 + t] = (t < 128) ? A[b * 128 + t] : B[b * 128 + (t - 128)];
}

__global__ __launch_bounds__(256) void add3_kernel(
    const float* __restrict__ a, const float* __restrict__ b,
    const float* __restrict__ c, float* __restrict__ y, long n) {
  long i = (long)blockIdx.x * 256 + threadIdx.x;
  if (i < n) y[i] = a[i] + b[i] + c[i];
}

// ------------------------- host orchestration ------------------------------
extern "C" void kernel_launch(void* const* d_in, const int* in_sizes, int n_in,
                              void* d_out, int out_size, void* d_ws,
                              size_t ws_size, hipStream_t stream) {
  (void)in_sizes; (void)n_in; (void)out_size; (void)ws_size;
  auto F = [&](int i) { return (const float*)d_in[i]; };
  auto I = [&](int i) { return (const int*)d_in[i]; };

  const int *nodes_u = I(0), *nodes_i = I(1);
  const int *hist_u = I(2), *hist_ur = I(3);
  const int *hist_i = I(4), *hist_ir = I(5);
  const int *social = I(6);
  const float *u2e = F(7), *i2e = F(8), *r2e = F(9);
  // param flatten (depth-first, dict insertion order):
  // enc_u: att0@10, att1@18, gate@26, gate1@28, mlp@30
  // enc_i: att0@38, att1@46, gate@54, gate1@56, mlp@58
  // soc:   att0@66, att1@74, gate@82, mlp@84
  // user_agg@92, user_agg_list0@100, item_agg0@108, item_agg1@116
  // predict@124, predict_bn@126, final@128

  char* wp = (char*)d_ws;
  auto alloc = [&](size_t nbytes) -> void* {
    void* p = wp;
    wp += (nbytes + 255) & ~(size_t)255;
    return p;
  };
  float* O     = (float*)alloc((size_t)MROWS * 128 * 4);  // o / o_n in place
  float* H1    = (float*)alloc((size_t)MROWS * 128 * 4);
  float* H2    = (float*)alloc((size_t)MROWS * 32 * 4);
  float* Sb    = (float*)alloc((size_t)MROWS * 4);
  float* Alp   = (float*)alloc((size_t)MROWS * 4);
  float* Wgt   = (float*)alloc((size_t)BATCH * 64 * 4);
  float* feats = (float*)alloc((size_t)BATCH * 128 * 4);
  float* repR  = (float*)alloc((size_t)BATCH * 128 * 4);
  float* repN  = (float*)alloc((size_t)BATCH * 128 * 4);
  float* selfF = (float*)alloc((size_t)BATCH * 128 * 4);
  float* selfN = (float*)alloc((size_t)BATCH * 128 * 4);
  float* neigh = (float*)alloc((size_t)BATCH * 128 * 4);
  float* itemS = (float*)alloc((size_t)BATCH * 128 * 4);
  float* socS  = (float*)alloc((size_t)BATCH * 128 * 4);
  float* uB    = (float*)alloc((size_t)BATCH * 128 * 4);
  float* vB    = (float*)alloc((size_t)BATCH * 128 * 4);
  float* g0    = (float*)alloc((size_t)BATCH * 128 * 4);
  float* cat   = (float*)alloc((size_t)BATCH * 256 * 4);
  float* tA    = (float*)alloc((size_t)BATCH * 256 * 4);
  float* tB2   = (float*)alloc((size_t)BATCH * 128 * 4);
  float* mu    = (float*)alloc(256 * 4);
  float* var   = (float*)alloc(256 * 4);

  // ---- weight pre-pass: transposed f16 copies (tiny, L2-resident)
  auto cvtW = [&](int idx, int K, int N) -> const _Float16* {
    _Float16* p = (_Float16*)alloc((size_t)K * N * 2);
    int tot = K * N;
    wtrans_kernel<<<(tot + 255) / 256, 256, 0, stream>>>(F(idx), p, K, N);
    return p;
  };
  // attention weights: att1(256x128)@+0, att2(128x32)@+2 for each head base
  const _Float16 *eu_a1_0 = cvtW(10, 256, 128), *eu_a2_0 = cvtW(12, 128, 32);
  const _Float16 *eu_a1_1 = cvtW(18, 256, 128), *eu_a2_1 = cvtW(20, 128, 32);
  const _Float16 *ei_a1_0 = cvtW(38, 256, 128), *ei_a2_0 = cvtW(40, 128, 32);
  const _Float16 *ei_a1_1 = cvtW(46, 256, 128), *ei_a2_1 = cvtW(48, 128, 32);
  const _Float16 *so_a1_0 = cvtW(66, 256, 128), *so_a2_0 = cvtW(68, 128, 32);
  const _Float16 *so_a1_1 = cvtW(74, 256, 128), *so_a2_1 = cvtW(76, 128, 32);
  const _Float16 *eu_gate  = cvtW(26, 384, 128), *eu_gate1 = cvtW(28, 384, 128);
  const _Float16 *ei_gate  = cvtW(54, 384, 128), *ei_gate1 = cvtW(56, 384, 128);
  const _Float16 *so_gate  = cvtW(82, 384, 128);
  const _Float16 *eu_m1 = cvtW(32, 128, 128), *eu_m2 = cvtW(36, 128, 128);
  const _Float16 *ei_m1 = cvtW(60, 128, 128), *ei_m2 = cvtW(64, 128, 128);
  const _Float16 *so_m1 = cvtW(86, 128, 128), *so_m2 = cvtW(90, 128, 128);
  const _Float16 *ua_m1 = cvtW(94, 256, 128), *ua_m2 = cvtW(98, 128, 128);
  const _Float16 *ul_m1 = cvtW(102, 128, 128), *ul_m2 = cvtW(106, 128, 128);
  const _Float16 *i0_m1 = cvtW(110, 128, 128), *i0_m2 = cvtW(114, 128, 128);
  const _Float16 *i1_m1 = cvtW(118, 128, 128), *i1_m2 = cvtW(122, 128, 128);
  const _Float16 *pred  = cvtW(124, 384, 128);

  const dim3 blk(128);
#define GNULLS nullptr, nullptr, nullptr, nullptr, nullptr

  // ---- two attention heads over On (normalized o), rep0 = normalized self
  auto run_heads = [&](const float* On, const float* rep0N, int ab0, int ab1,
                       const _Float16* a1t0, const _Float16* a2t0,
                       const _Float16* a1t1, const _Float16* a2t1) {
    const float* rep = rep0N;
    int abs_[2] = {ab0, ab1};
    const _Float16* a1t[2] = {a1t0, a1t1};
    const _Float16* a2t[2] = {a2t0, a2t1};
    for (int h = 0; h < 2; ++h) {
      int ab = abs_[h];
      rowdot_kernel<128, 1><<<MROWS / 4, blk, 0, stream>>>(On, F(ab + 6),
                                                           F(ab + 7), Alp, MROWS);
      gemm_wmma_kernel<A_CONCAT_BCAST, E_SELU, 256>
          <<<dim3(MROWS / 64, 8), blk, 0, stream>>>(
              On, rep, a1t[h], F(ab + 1), H1, GNULLS, nullptr, nullptr,
              MROWS, 128);
      gemm_wmma_kernel<A_PLAIN, E_SELU, 128>
          <<<dim3(MROWS / 64, 2), blk, 0, stream>>>(
              H1, nullptr, a2t[h], F(ab + 3), H2, GNULLS, nullptr, nullptr,
              MROWS, 32);
      rowdot_kernel<32, 0><<<MROWS / 4, blk, 0, stream>>>(H2, F(ab + 4),
                                                          F(ab + 5), Sb, MROWS);
      entmax_kernel<<<BATCH, 32, 0, stream>>>(Sb, Alp, Wgt);
      wsum_kernel<<<BATCH, blk, 0, stream>>>(Wgt, On, repR, feats, h == 0);
      if (h == 0) {
        l2norm_kernel<<<BATCH, blk, 0, stream>>>(repR, repN);
        rep = repN;
      }
    }
  };

  // ---- MLP: bn0 -> selu(lin_in) -> bn1 -> lin_out
  auto run_mlp = [&](const float* X, int Cin, float* Y, int base,
                     float prescale, const _Float16* w1t, const _Float16* w2t) {
    bn_stats_kernel<<<Cin, 256, 0, stream>>>(X, mu, var, BATCH, Cin, prescale);
    long tot = (long)BATCH * Cin;
    bn_apply_kernel<<<(tot + 255) / 256, 256, 0, stream>>>(
        X, tA, mu, var, F(base), F(base + 1), tot, Cin, prescale);
    if (Cin == 256)
      gemm_wmma_kernel<A_PLAIN, E_SELU, 256>
          <<<dim3(BATCH / 64, 8), blk, 0, stream>>>(
              tA, nullptr, w1t, F(base + 3), tB2, GNULLS, nullptr, nullptr,
              BATCH, 128);
    else
      gemm_wmma_kernel<A_PLAIN, E_SELU, 128>
          <<<dim3(BATCH / 64, 8), blk, 0, stream>>>(
              tA, nullptr, w1t, F(base + 3), tB2, GNULLS, nullptr, nullptr,
              BATCH, 128);
    bn_stats_kernel<<<128, 256, 0, stream>>>(tB2, mu, var, BATCH, 128, 1.f);
    tot = (long)BATCH * 128;
    bn_apply_kernel<<<(tot + 255) / 256, 256, 0, stream>>>(
        tB2, tA, mu, var, F(base + 4), F(base + 5), tot, 128, 1.f);
    gemm_wmma_kernel<A_PLAIN, E_NONE, 128>
        <<<dim3(BATCH / 64, 8), blk, 0, stream>>>(
            tA, nullptr, w2t, F(base + 7), Y, GNULLS, nullptr, nullptr,
            BATCH, 128);
  };

  // ---- user/item aggregation over rated history (gate + attention + gate1)
  auto run_enc = [&](const int* nodes, const int* htab, const int* rtab,
                     const float* embSelf, const float* embT, int base,
                     const _Float16* gateT, const _Float16* gate1T,
                     const _Float16* a1t0, const _Float16* a2t0,
                     const _Float16* a1t1, const _Float16* a2t1,
                     const _Float16* m1t, const _Float16* m2t, float* OUT) {
    int gateB = base + 16, gate1B = base + 18, mlpB = base + 20;
    gemm_wmma_kernel<A_GATE_GATHER, E_GATE_O, 384>
        <<<dim3(MROWS / 64, 8), blk, 0, stream>>>(
            nullptr, nullptr, gateT, F(gateB + 1), O, nodes, htab, rtab,
            embT, r2e, nullptr, nullptr, MROWS, 128);
    l2norm_kernel<<<MROWS, blk, 0, stream>>>(O, O);   // o -> o_n in place
    gather_self_kernel<<<BATCH, blk, 0, stream>>>(nodes, embSelf, selfF);
    l2norm_kernel<<<BATCH, blk, 0, stream>>>(selfF, selfN);
    run_heads(O, selfN, base, base + 8, a1t0, a2t0, a1t1, a2t1);
    run_mlp(feats, 128, neigh, mlpB, 0.5f, m1t, m2t);  // mean of 2 heads
    gemm_wmma_kernel<A_CONCAT3, E_COMB, 384>
        <<<dim3(BATCH / 64, 8), blk, 0, stream>>>(
            selfF, neigh, gate1T, F(gate1B + 1), OUT, GNULLS, selfF, neigh,
            BATCH, 128);
  };

  // ================= pipeline =================
  // item-space aggregation for users
  run_enc(nodes_u, hist_u, hist_ur, u2e, i2e, 10, eu_gate, eu_gate1,
          eu_a1_0, eu_a2_0, eu_a1_1, eu_a2_1, eu_m1, eu_m2, itemS);

  // social-space aggregation
  gather_rows_kernel<<<MROWS, blk, 0, stream>>>(nodes_u, social, u2e, O);
  l2norm_kernel<<<MROWS, blk, 0, stream>>>(O, O);
  gather_self_kernel<<<BATCH, blk, 0, stream>>>(nodes_u, u2e, selfF);
  l2norm_kernel<<<BATCH, blk, 0, stream>>>(selfF, selfN);
  run_heads(O, selfN, 66, 74, so_a1_0, so_a2_0, so_a1_1, so_a2_1);
  run_mlp(feats, 128, neigh, 84, 0.5f, so_m1, so_m2);
  gemm_wmma_kernel<A_CONCAT3, E_COMB, 384>
      <<<dim3(BATCH / 64, 8), blk, 0, stream>>>(
          selfF, neigh, so_gate, F(83), socS, GNULLS, selfF, neigh,
          BATCH, 128);

  // u = mlp(user_agg, [item|social]) + item + social ; then user_agg_list
  concat2_kernel<<<BATCH, 256, 0, stream>>>(itemS, socS, cat);
  run_mlp(cat, 256, g0, 92, 1.f, ua_m1, ua_m2);
  add3_kernel<<<((long)BATCH * 128 + 255) / 256, 256, 0, stream>>>(
      g0, itemS, socS, uB, (long)BATCH * 128);
  run_mlp(uB, 128, g0, 100, 1.f, ul_m1, ul_m2);       // u = g0

  // item side
  run_enc(nodes_i, hist_i, hist_ir, i2e, u2e, 38, ei_gate, ei_gate1,
          ei_a1_0, ei_a2_0, ei_a1_1, ei_a2_1, ei_m1, ei_m2, vB);
  run_mlp(vB, 128, socS, 108, 1.f, i0_m1, i0_m2);     // reuse socS
  run_mlp(socS, 128, vB, 116, 1.f, i1_m1, i1_m2);     // v = vB

  // predict: eps = sigmoid([u|v|u*v]@Wp+b); lat = eps*u + (1-eps)*v
  gemm_wmma_kernel<A_CONCAT3, E_COMB, 384>
      <<<dim3(BATCH / 64, 8), blk, 0, stream>>>(
          g0, vB, pred, F(125), itemS /*lat*/, GNULLS, g0, vB, BATCH, 128);
  bn_stats_kernel<<<128, 256, 0, stream>>>(itemS, mu, var, BATCH, 128, 1.f);
  bn_apply_kernel<<<((long)BATCH * 128 + 255) / 256, 256, 0, stream>>>(
      itemS, tA, mu, var, F(126), F(127), (long)BATCH * 128, 128, 1.f);
  rowdot_kernel<128, 0><<<BATCH / 4, blk, 0, stream>>>(tA, F(128), F(129),
                                                       (float*)d_out, BATCH);
#undef GNULLS
}
